// CausalSelfAttention_9431748182072
// MI455X (gfx1250) — matrix operand order repaired and checked by reading, not compiled, not run on hardware
//
#include <hip/hip_runtime.h>

typedef __attribute__((ext_vector_type(16))) __bf16 v16bf;
typedef __attribute__((ext_vector_type(2)))  __bf16 v2bf;
typedef __attribute__((ext_vector_type(8)))  float  v8f;

// Native converts: backend lowers these to v_cvt_pk_bf16_f32.
static __device__ __forceinline__ __bf16 f2bf(float f) { return (__bf16)f; }
static __device__ __forceinline__ unsigned pack2bf(float lo, float hi) {
  union { v2bf v; unsigned u; } t;
  t.v[0] = (__bf16)lo;
  t.v[1] = (__bf16)hi;
  return t.u;
}

static __device__ __forceinline__ v8f zero_v8f() {
  v8f z;
#pragma unroll
  for (int i = 0; i < 8; ++i) z[i] = 0.0f;
  return z;
}

// A-layout fragment: two 16B chunks at base and base+8 elements
static __device__ __forceinline__ v16bf load_afrag(const __bf16* base) {
  union { uint4 u[2]; v16bf v; } t;
  t.u[0] = *(const uint4*)(base);
  t.u[1] = *(const uint4*)(base + 8);
  return t.v;
}
// 16 contiguous bf16 (K fragments: k-dim contiguous in memory)
static __device__ __forceinline__ v16bf load_cfrag(const __bf16* p) {
  union { uint4 u[2]; v16bf v; } t;
  t.u[0] = *(const uint4*)(p);
  t.u[1] = *(const uint4*)(p + 8);
  return t.v;
}
// 8 contiguous k-pair-packed dwords (Bp / Vp fragments)
static __device__ __forceinline__ v16bf load_pfrag(const unsigned* p) {
  union { uint4 u[2]; v16bf v; } t;
  t.u[0] = *(const uint4*)(p);
  t.u[1] = *(const uint4*)(p + 4);
  return t.v;
}

// ---------------------------------------------------------------------------
// C(f32)[MxN] = A(f32)[MxK] @ B(f32)[KxN], bf16 WMMA compute.
// N,K are template constants so strided global accesses fold into
// base + immediate-offset (24-bit IOFFSET) clause'd loads/stores.
// 128x128 block tile, 256 threads (8 waves), wave tile 32x64 (2x4 WMMAs), BK=32.
// ---------------------------------------------------------------------------
#define GM_BM 128
#define GM_BN 128
#define GM_BK 32

template <int N, int K>
__global__ __launch_bounds__(256) void gemm_bf16_wmma(
    const float* __restrict__ A, const float* __restrict__ B,
    float* __restrict__ C) {
  __shared__ __align__(16) __bf16   As[GM_BM][GM_BK + 8];   // 80B rows
  __shared__ __align__(16) unsigned Bp[GM_BN][20];          // k-pair packed, 80B rows

  const int tid  = threadIdx.x;
  const int wave = tid >> 5;
  const int lane = tid & 31;
  const int hf   = lane >> 4;
  const int lm   = lane & 15;

  const int bm = blockIdx.x * GM_BM;
  const int bn = blockIdx.y * GM_BN;
  const int wm = (wave >> 1) * 32;   // 0,32,64,96
  const int wn = (wave & 1) * 64;    // 0,64

  v8f acc[2][4];
#pragma unroll
  for (int ti = 0; ti < 2; ++ti)
#pragma unroll
    for (int tj = 0; tj < 4; ++tj) acc[ti][tj] = zero_v8f();

  const int ar = tid >> 1, ac = (tid & 1) * 16;       // A coop-load coords
  const int bcol = tid & 127, bkw0 = (tid >> 7) * 8;  // B coop-load coords

  const float* srcA0 = A + (size_t)(bm + ar) * K + ac;
  const float* srcB0 = B + (size_t)(2 * bkw0) * N + bn + bcol;

  for (int k0 = 0; k0 < K; k0 += GM_BK) {
    // ---- A tile: 4x b128 loads -> 8 packed cvts -> 2x ds_store_b128 ----
    {
      const float* srcA = srcA0 + k0;
      unsigned w[8];
#pragma unroll
      for (int i = 0; i < 8; ++i) w[i] = pack2bf(srcA[2 * i], srcA[2 * i + 1]);
      uint4* dst = (uint4*)&As[ar][ac];
      dst[0] = make_uint4(w[0], w[1], w[2], w[3]);
      dst[1] = make_uint4(w[4], w[5], w[6], w[7]);
    }
    // ---- B tile: k-pair packed column-major; const strides -> imm offsets ----
    {
      const float* srcB = srcB0 + (size_t)k0 * N;
#pragma unroll
      for (int i = 0; i < 8; ++i) {
        const float f0 = srcB[(size_t)(2 * i) * N];
        const float f1 = srcB[(size_t)(2 * i + 1) * N];
        Bp[bcol][bkw0 + i] = pack2bf(f0, f1);
      }
    }
    if (k0 + GM_BK < K) {   // gfx1250 global_prefetch of next tiles
      __builtin_prefetch(srcA0 + k0 + GM_BK, 0, 1);
      __builtin_prefetch(srcB0 + (size_t)(k0 + GM_BK) * N, 0, 1);
    }
    __syncthreads();

    v16bf afrag[2];
#pragma unroll
    for (int ti = 0; ti < 2; ++ti)
      afrag[ti] = load_afrag(&As[wm + ti * 16 + lm][hf * 16]);
    v16bf bfrag[4];
#pragma unroll
    for (int tj = 0; tj < 4; ++tj)
      bfrag[tj] = load_pfrag(&Bp[wn + tj * 16 + lm][hf * 8]);

#pragma unroll
    for (int ti = 0; ti < 2; ++ti)
#pragma unroll
      for (int tj = 0; tj < 4; ++tj)
        acc[ti][tj] = __builtin_amdgcn_wmma_f32_16x16x32_bf16(
            false, afrag[ti], false, bfrag[tj], (short)0, acc[ti][tj], false, false);

    __syncthreads();
  }

  // ---- epilogue: one per-lane base + immediate-offset NT stores ----
  float* cbase = C + (size_t)(bm + wm + 8 * hf) * N + (bn + wn + lm);
#pragma unroll
  for (int ti = 0; ti < 2; ++ti)
#pragma unroll
    for (int tj = 0; tj < 4; ++tj)
#pragma unroll
      for (int r = 0; r < 8; ++r)
        __builtin_nontemporal_store(acc[ti][tj][r],
                                    cbase + (size_t)(ti * 16 + r) * N + tj * 16);
}

// ---------------------------------------------------------------------------
// RoPE + repack: qkv f32 [B*T, 3C] -> Qb/Kb/Vb bf16 [B,H,T,D].
// softmax scale * log2(e) folded into Q: downstream softmax runs in exp2 domain.
// ---------------------------------------------------------------------------
__global__ __launch_bounds__(256) void rope_pack(
    const float* __restrict__ qkv,
    __bf16* __restrict__ Qb, __bf16* __restrict__ Kb, __bf16* __restrict__ Vb) {
  const int T = 2048, C = 2048, H = 16, D = 128;
  const size_t idx = (size_t)blockIdx.x * 256 + threadIdx.x;
  const int i = (int)(idx & 63);
  const int h = (int)((idx >> 6) & 15);
  const int t = (int)((idx >> 10) & 2047);
  const int b = (int)(idx >> 21);

  const float* row = qkv + ((size_t)b * T + t) * (3 * C);
  const float invf = __expf((float)i * -0.14391156831212787f); // 10000^(-2i/128)
  const float ang = (float)t * invf;
  const float c = __cosf(ang), s = __sinf(ang);
  const float scale = 0.12751744577734657f;  // (1/sqrt(128)) * log2(e)

  const int base = h * D + 2 * i;
  const float q1 = row[base],         q2 = row[base + 1];
  const float k1 = row[C + base],     k2 = row[C + base + 1];
  const float v1 = row[2 * C + base], v2 = row[2 * C + base + 1];

  const size_t o = (((size_t)(b * H + h) * T) + t) * D + 2 * i;
  *(unsigned*)(Qb + o) = pack2bf((q1 * c - q2 * s) * scale, (q1 * s + q2 * c) * scale);
  *(unsigned*)(Kb + o) = pack2bf(k1 * c - k2 * s, k1 * s + k2 * c);
  *(unsigned*)(Vb + o) = pack2bf(v1, v2);
}

// ---------------------------------------------------------------------------
// Causal flash attention, bf16 WMMA, exp2-domain streaming softmax.
// Grid: x = T/64 query blocks, y = B*H. 128 threads = 4 waves, 16 q-rows/wave.
// ---------------------------------------------------------------------------
#define AT_BM 64
#define AT_BN 64

__global__ __launch_bounds__(128) void attn_fwd(
    const __bf16* __restrict__ Qb, const __bf16* __restrict__ Kb,
    const __bf16* __restrict__ Vb, float* __restrict__ Y) {
  const int T = 2048, D = 128, H = 16;
  const int bh = blockIdx.y;
  const int b = bh >> 4, h = bh & 15;
  const int qblk = blockIdx.x;

  const int tid  = threadIdx.x;
  const int wave = tid >> 5;
  const int lane = tid & 31;
  const int hf   = lane >> 4;
  const int lm   = lane & 15;

  const __bf16* Q = Qb + (size_t)bh * T * D;
  const __bf16* K = Kb + (size_t)bh * T * D;
  const __bf16* V = Vb + (size_t)bh * T * D;

  __shared__ __align__(16) __bf16   Ks[AT_BN][D + 8];     // 272B rows
  __shared__ __align__(16) unsigned Vp[D][36];            // k-pair packed V
  __shared__ __align__(16) __bf16   Ps[4][16][AT_BN + 8]; // per-wave P staging

  const int qrow0 = qblk * AT_BM + wave * 16;

  v16bf qf[4];
#pragma unroll
  for (int kt = 0; kt < 4; ++kt)
    qf[kt] = load_afrag(Q + (size_t)(qrow0 + lm) * D + kt * 32 + hf * 16);

  const float NEG_INF = -__builtin_inff();
  float m_i[8], l_i[8];
  v8f o_acc[8];
#pragma unroll
  for (int r = 0; r < 8; ++r) { m_i[r] = NEG_INF; l_i[r] = 0.0f; }
#pragma unroll
  for (int dt = 0; dt < 8; ++dt) o_acc[dt] = zero_v8f();

  const int jmax = (qblk + 1) * AT_BM;           // causal bound (uniform per block)
  const int kr = tid >> 1, kc = (tid & 1) * 64;  // K coop-load coords
  const int vd0 = (tid & 63) * 2;                // V pack: d pair
  const int vkw0 = (tid >> 6) * 16;              //         kw range

  for (int j0 = 0; j0 < jmax; j0 += AT_BN) {
    // ---- K tile: bulk 16B copies ----
    {
      const uint4* srcK = (const uint4*)(K + (size_t)(j0 + kr) * D + kc);
      uint4* dstK = (uint4*)(&Ks[kr][kc]);
#pragma unroll
      for (int i = 0; i < 8; ++i) dstK[i] = srcK[i];
    }
    // ---- V tile: k-pair packed, dword loads + 2x2 in-register transpose ----
    {
      const __bf16* srcV = V + (size_t)(j0 + 2 * vkw0) * D + vd0;
#pragma unroll
      for (int i = 0; i < 16; ++i) {
        const unsigned u0 = *(const unsigned*)(srcV + (size_t)(2 * i) * D);
        const unsigned u1 = *(const unsigned*)(srcV + (size_t)(2 * i + 1) * D);
        Vp[vd0][vkw0 + i]     = (u0 & 0xffffu) | (u1 << 16);
        Vp[vd0 + 1][vkw0 + i] = (u0 >> 16) | (u1 & 0xffff0000u);
      }
    }
    __syncthreads();

    // ---- S = Q @ K^T (4 N-tiles x 4 K-steps); S is in log2 domain ----
    v8f s[4];
#pragma unroll
    for (int nt = 0; nt < 4; ++nt) s[nt] = zero_v8f();
#pragma unroll
    for (int nt = 0; nt < 4; ++nt) {
#pragma unroll
      for (int kt = 0; kt < 4; ++kt) {
        const v16bf bf = load_cfrag(&Ks[nt * 16 + lm][kt * 32 + hf * 16]);
        s[nt] = __builtin_amdgcn_wmma_f32_16x16x32_bf16(
            false, qf[kt], false, bf, (short)0, s[nt], false, false);
      }
    }

    // ---- causal mask + streaming softmax in exp2 domain ----
#pragma unroll
    for (int r = 0; r < 8; ++r) {
      const int grow = qrow0 + r + 8 * hf;
      float mx = m_i[r];
#pragma unroll
      for (int nt = 0; nt < 4; ++nt) {
        const int key = j0 + nt * 16 + lm;
        float val = s[nt][r];
        if (key > grow) val = NEG_INF;
        s[nt][r] = val;
        mx = fmaxf(mx, val);
      }
      mx = fmaxf(mx, __shfl_xor(mx, 1, 32));
      mx = fmaxf(mx, __shfl_xor(mx, 2, 32));
      mx = fmaxf(mx, __shfl_xor(mx, 4, 32));
      mx = fmaxf(mx, __shfl_xor(mx, 8, 32));

      const float corr = __builtin_amdgcn_exp2f(m_i[r] - mx);  // single v_exp_f32
      float rowsum = 0.0f;
#pragma unroll
      for (int nt = 0; nt < 4; ++nt) {
        const float p = __builtin_amdgcn_exp2f(s[nt][r] - mx);
        s[nt][r] = p;
        rowsum += p;
      }
      rowsum += __shfl_xor(rowsum, 1, 32);
      rowsum += __shfl_xor(rowsum, 2, 32);
      rowsum += __shfl_xor(rowsum, 4, 32);
      rowsum += __shfl_xor(rowsum, 8, 32);

      l_i[r] = l_i[r] * corr + rowsum;
      m_i[r] = mx;
#pragma unroll
      for (int dt = 0; dt < 8; ++dt) o_acc[dt][r] *= corr;

#pragma unroll
      for (int nt = 0; nt < 4; ++nt)
        Ps[wave][r + 8 * hf][nt * 16 + lm] = f2bf(s[nt][r]);
    }
    __syncthreads();   // P visible across lanes; Ks reads done

    // ---- O += P @ V (8 d-tiles x 2 K-steps over 64 keys) ----
    v16bf pf[2];
#pragma unroll
    for (int kt = 0; kt < 2; ++kt)
      pf[kt] = load_afrag(&Ps[wave][lm][kt * 32 + hf * 16]);
#pragma unroll
    for (int dt = 0; dt < 8; ++dt) {
#pragma unroll
      for (int kt = 0; kt < 2; ++kt) {
        const v16bf vf = load_pfrag(&Vp[dt * 16 + lm][kt * 16 + hf * 8]);
        o_acc[dt] = __builtin_amdgcn_wmma_f32_16x16x32_bf16(
            false, pf[kt], false, vf, (short)0, o_acc[dt], false, false);
      }
    }
    __syncthreads();   // Vp/Ks reads done before next block overwrites
  }

  // ---- normalize (v_rcp + muls) and NT-store Y[b, t, h*128+d] ----
  float inv_l[8];
#pragma unroll
  for (int r = 0; r < 8; ++r) inv_l[r] = __builtin_amdgcn_rcpf(l_i[r]);

  float* ybase = Y + ((size_t)b * T + qrow0 + 8 * hf) * (H * D) + h * D + lm;
#pragma unroll
  for (int dt = 0; dt < 8; ++dt) {
#pragma unroll
    for (int r = 0; r < 8; ++r) {
      const float val = o_acc[dt][r] * inv_l[r];
      __builtin_nontemporal_store(val, ybase + (size_t)r * (H * D) + dt * 16);
    }
  }
}

// ---------------------------------------------------------------------------
extern "C" void kernel_launch(void* const* d_in, const int* in_sizes, int n_in,
                              void* d_out, int out_size, void* d_ws, size_t ws_size,
                              hipStream_t stream) {
  const float* x      = (const float*)d_in[0];   // [4,2048,2048]
  const float* w_attn = (const float*)d_in[1];   // [2048,6144]
  const float* w_proj = (const float*)d_in[2];   // [2048,2048]
  float* out = (float*)d_out;                    // [4,2048,2048]

  char* ws = (char*)d_ws;
  const size_t M = 8192;                                   // B*T
  const size_t QKV_BYTES = M * 6144 * sizeof(float);       // 192 MB
  const size_t HT = (size_t)4 * 16 * 2048 * 128;           // elems per bf16 tensor

  float*  qkv = (float*)ws;
  __bf16* Qb  = (__bf16*)(ws + QKV_BYTES);
  __bf16* Kb  = Qb + HT;
  __bf16* Vb  = Kb + HT;
  float*  y   = (float*)ws;                      // reuse qkv region after RoPE

  gemm_bf16_wmma<6144, 2048><<<dim3(8192 / GM_BM, 6144 / GM_BN), 256, 0, stream>>>(
      x, w_attn, qkv);

  rope_pack<<<dim3((4 * 2048 * 16 * 64) / 256), 256, 0, stream>>>(qkv, Qb, Kb, Vb);

  attn_fwd<<<dim3(2048 / AT_BM, 4 * 16), 128, 0, stream>>>(Qb, Kb, Vb, y);

  gemm_bf16_wmma<2048, 2048><<<dim3(8192 / GM_BM, 2048 / GM_BN), 256, 0, stream>>>(
      y, w_proj, out);
}
